// VectorQuantizerEMA_84713934946735
// MI455X (gfx1250) — matrix operand order repaired and checked by reading, not compiled, not run on hardware
//
#include <hip/hip_runtime.h>
#include <math.h>

// Problem constants (match reference)
#define NUM_CODES 4096
#define CODE_DIM  256
#define BATCH     65536
#define KTILES    (NUM_CODES / 16)     // 256 k-tiles of 16 codes
#define DCHUNKS   (CODE_DIM / 32)      // 8 chunks of K=32 for bf16 WMMA

// packed embedding: per code row, 256 bf16-hi then 256 bf16-lo (512 u16 = 1KB)
#define ROW_U16   (2 * CODE_DIM)                 // 512
#define TILE_U16  (16 * ROW_U16)                 // 8192 elements = 16KB
#define TOT_U16   ((unsigned)NUM_CODES * ROW_U16) // 2097152 elements

typedef __bf16 bf16_t;
typedef __attribute__((ext_vector_type(16))) bf16_t         v16bf;
typedef __attribute__((ext_vector_type(16))) unsigned short v16u;
typedef __attribute__((ext_vector_type(8)))  unsigned short v8u;
typedef __attribute__((ext_vector_type(8)))  float          v8f;
typedef __attribute__((ext_vector_type(4)))  float          f32x4;
typedef __attribute__((ext_vector_type(4)))  unsigned int   u32x4;
typedef __attribute__((ext_vector_type(8)))  unsigned int   u32x8;

__device__ __forceinline__ unsigned short f32_to_bf16(float x) {
  unsigned int u = __builtin_bit_cast(unsigned int, x);
  u += 0x7FFFu + ((u >> 16) & 1u);   // round to nearest even
  return (unsigned short)(u >> 16);
}
__device__ __forceinline__ float bf16_to_f32(unsigned short h) {
  unsigned int u = ((unsigned int)h) << 16;
  return __builtin_bit_cast(float, u);
}

// B-fragment (32x16 bf16) from LDS: lane L holds code n = n0 + L%16,
// K {8*half..8*half+7, 16+8*half..16+8*half+7} (two ds_load_b128).
__device__ __forceinline__ v16bf load_bfrag(const unsigned short* p) {
  v8u a = *(const v8u*)p;
  v8u b = *(const v8u*)(p + 16);
  v16u u;
#pragma unroll
  for (int i = 0; i < 8; ++i) { u[i] = a[i]; u[i + 8] = b[i]; }
  return __builtin_bit_cast(v16bf, u);
}

// One 16x16 output tile: dot(z_rows, e_codes) over D=256, bf16 hi/lo split:
// acc += Ahi*Bhi + Alo*Bhi + Ahi*Blo   (24 v_wmma_f32_16x16x32_bf16)
__device__ __forceinline__ v8f tile_acc(const v16bf* a_hi, const v16bf* a_lo,
                                        const unsigned short* bh,
                                        const unsigned short* bl) {
  v8f acc = {};
#pragma unroll
  for (int c = 0; c < DCHUNKS; ++c) {
    v16bf h = load_bfrag(bh + c * 32);
    v16bf l = load_bfrag(bl + c * 32);
    acc = __builtin_amdgcn_wmma_f32_16x16x32_bf16(false, a_hi[c], false, h, (short)0, acc, false, false);
    acc = __builtin_amdgcn_wmma_f32_16x16x32_bf16(false, a_lo[c], false, h, (short)0, acc, false, false);
    acc = __builtin_amdgcn_wmma_f32_16x16x32_bf16(false, a_hi[c], false, l, (short)0, acc, false, false);
  }
  return acc;
}

// ---------------- prep: embedding -> packed bf16 hi|lo + ||e||^2 ----------------
__global__ void vq_prep(const float* __restrict__ emb,
                        unsigned short* __restrict__ e_pack,
                        float* __restrict__ e_norm) {
  const int row = blockIdx.x;            // one code row per block, 256 threads
  const int t = threadIdx.x;
  float x = emb[(size_t)row * CODE_DIM + t];
  unsigned short h = f32_to_bf16(x);
  e_pack[(size_t)row * ROW_U16 + t] = h;
  e_pack[(size_t)row * ROW_U16 + CODE_DIM + t] = f32_to_bf16(x - bf16_to_f32(h));
  float p = x * x;
#pragma unroll
  for (int off = 16; off > 0; off >>= 1) p += __shfl_xor(p, off);
  __shared__ float w[8];
  if ((t & 31) == 0) w[t >> 5] = p;
  __syncthreads();
  if (t == 0) {
    float s = 0.f;
#pragma unroll
    for (int i = 0; i < 8; ++i) s += w[i];
    e_norm[row] = s;
  }
}

__global__ void vq_init(float* __restrict__ p, int n) {
  int i = blockIdx.x * blockDim.x + threadIdx.x;
  if (i < n) p[i] = 0.f;
}

// ---------------- main: TDM-staged tiles, WMMA distances, argmin + softmax ----------------
__global__ void __launch_bounds__(128)
vq_main(const float* __restrict__ z, const float* __restrict__ emb,
        const unsigned short* __restrict__ e_pack,
        const float* __restrict__ e_norm,
        float* __restrict__ g_avg, float* __restrict__ g_bin,
        float* __restrict__ g_scal,
        float* __restrict__ out_zq, float* __restrict__ out_idx) {
  __shared__ float s_enorm[NUM_CODES];
  __shared__ float s_avg[NUM_CODES];
  __shared__ unsigned short s_tile[2][TILE_U16];   // double-buffered 16KB tiles
  __shared__ float s_zz[4][16];
  __shared__ float s_m[4][16];
  __shared__ float s_rinv[4][16];
  __shared__ int   s_idx[4][16];

  for (int j = threadIdx.x; j < NUM_CODES; j += 128) {
    s_enorm[j] = e_norm[j];
    s_avg[j] = 0.f;
  }

  const int wave = threadIdx.x >> 5;
  const int lane = threadIdx.x & 31;
  const int half = lane >> 4;
  const int lm   = lane & 15;
  const int r0   = (blockIdx.x * 4 + wave) * 16;   // 16 z-rows per wave
  const int myrow = r0 + lm;

  // Tensor DMA descriptor group 1 (constant): 2B elements, 1-D copy of
  // tile_dim0 = 8192 elements (one 16-code tile) out of tensor_dim0 = 2M.
  u32x8 g1;
  g1[0] = 0x00010000u;                          // data_size=1 (2 bytes)
  g1[1] = (TOT_U16 & 0xFFFFu) << 16;            // tensor_dim0[15:0]
  g1[2] = ((TOT_U16 >> 16) & 0xFFFFu) | (1u << 16); // dim0[31:16]; tensor_dim1=1
  g1[3] = ((unsigned)TILE_U16) << 16;           // tensor_dim1 hi=0; tile_dim0=8192
  g1[4] = 0u;                                   // tile_dim1=0, tile_dim2=0
  g1[5] = TOT_U16;                              // tensor_dim0_stride[31:0]
  g1[6] = 0u;
  g1[7] = 0u;

  auto issue_tdm = [&](int t, int b) {
    unsigned long long ga =
        (unsigned long long)(uintptr_t)(e_pack + (size_t)t * TILE_U16);
    u32x4 g0;
    g0[0] = 1u;                                         // count=1 descriptor
    g0[1] = (unsigned)(uintptr_t)(&s_tile[b][0]);       // lds_addr
    g0[2] = (unsigned)ga;                               // global_addr[31:0]
    g0[3] = ((unsigned)(ga >> 32) & 0x01FFFFFFu) | (2u << 30); // addr[56:32]|type=2
    asm volatile("tensor_load_to_lds %0, %1" : : "s"(g0), "s"(g1) : "memory");
  };

  // ---- load 16 z rows as A fragments (bf16 hi/lo), compute ||z||^2 ----
  // 16-bit A layout: half 0 -> K {0..7,16..23}, half 1 -> K {8..15,24..31}
  v16bf a_hi[DCHUNKS], a_lo[DCHUNKS];
  float zzp = 0.f;
  const float* zrow = z + (size_t)myrow * CODE_DIM;
#pragma unroll
  for (int c = 0; c < DCHUNKS; ++c) {
    const float* p0 = zrow + c * 32 + half * 8;
    f32x4 t0 = *(const f32x4*)(p0);
    f32x4 t1 = *(const f32x4*)(p0 + 4);
    f32x4 t2 = *(const f32x4*)(p0 + 16);
    f32x4 t3 = *(const f32x4*)(p0 + 20);
    float xs[16];
#pragma unroll
    for (int i = 0; i < 4; ++i) {
      xs[i] = t0[i]; xs[4 + i] = t1[i]; xs[8 + i] = t2[i]; xs[12 + i] = t3[i];
    }
    v16u uh, ul;
#pragma unroll
    for (int i = 0; i < 16; ++i) {
      float x = xs[i];
      zzp += x * x;
      unsigned short h = f32_to_bf16(x);
      uh[i] = h;
      ul[i] = f32_to_bf16(x - bf16_to_f32(h));
    }
    a_hi[c] = __builtin_bit_cast(v16bf, uh);
    a_lo[c] = __builtin_bit_cast(v16bf, ul);
  }
  {
    float zztot = zzp + __shfl_xor(zzp, 16);
    if (half == 0) s_zz[wave][lm] = zztot;
  }

  // Double-buffered tile loop: wave 0 drives the TDM, all 4 waves consume
  // the staged tile from LDS. consume(acc, n) sees slot v -> row M=v+8*half,
  // code n = t*16 + lane%16 (C-matrix layout).
  auto tile_loop = [&](auto&& consume) {
    if (wave == 0) {
      issue_tdm(0, 0);
      __builtin_amdgcn_s_wait_tensorcnt(0);
    }
    __syncthreads();
    for (int t = 0; t < KTILES; ++t) {
      const int b = t & 1;
      if (wave == 0 && t + 1 < KTILES) issue_tdm(t + 1, b ^ 1);
      const unsigned short* bh = &s_tile[b][lm * ROW_U16 + half * 8];
      v8f acc = tile_acc(a_hi, a_lo, bh, bh + CODE_DIM);
      consume(acc, t * 16 + lm);
      if (wave == 0 && t + 1 < KTILES) __builtin_amdgcn_s_wait_tensorcnt(0);
      __syncthreads();
    }
  };

  // ---- pass 1: online (max, sum-exp, argmax) of score = 2*z.e - ||e||^2 ----
  float m[8], s[8];
  int bi[8];
#pragma unroll
  for (int v = 0; v < 8; ++v) { m[v] = -3.0e38f; s[v] = 0.f; bi[v] = 0; }

  tile_loop([&](const v8f& acc, int n) {
    float een = s_enorm[n];
#pragma unroll
    for (int v = 0; v < 8; ++v) {
      float score = 2.f * acc[v] - een;
      if (score > m[v]) {
        s[v] = s[v] * __expf(m[v] - score) + 1.f;
        m[v] = score;
        bi[v] = n;
      } else {
        s[v] += __expf(score - m[v]);
      }
    }
  });

  // reduce (m,s,argmax) across the 16 lanes of each half
#pragma unroll
  for (int v = 0; v < 8; ++v) {
#pragma unroll
    for (int off = 1; off < 16; off <<= 1) {
      float m2 = __shfl_xor(m[v], off);
      float s2 = __shfl_xor(s[v], off);
      int   i2 = __shfl_xor(bi[v], off);
      float mm = fmaxf(m[v], m2);
      s[v] = s[v] * __expf(m[v] - mm) + s2 * __expf(m2 - mm);
      if (m2 > m[v] || (m2 == m[v] && i2 < bi[v])) bi[v] = i2;  // first-min tie-break
      m[v] = mm;
    }
  }

  if (lm == 0) {
    float sse = 0.f;
#pragma unroll
    for (int v = 0; v < 8; ++v) {
      const int M = v + 8 * half;
      s_m[wave][M] = m[v];
      s_rinv[wave][M] = 1.f / s[v];
      s_idx[wave][M] = bi[v];
      out_idx[r0 + M] = (float)bi[v];
      atomicAdd(&g_bin[bi[v]], 1.f);
      sse += s_zz[wave][M] - m[v];   // d_min = ||z||^2 - max_score = ||z - z_q||^2
    }
    atomicAdd(&g_scal[0], sse);
  }

  // z_q_st rows = embedding[argmin]  (straight-through value == z_q)
#pragma unroll 1
  for (int r = 0; r < 16; ++r) {
    const int code = s_idx[wave][r];
    const f32x4* src = (const f32x4*)(emb + (size_t)code * CODE_DIM);
    f32x4* dst = (f32x4*)(out_zq + (size_t)(r0 + r) * CODE_DIM);
    for (int j = lane; j < CODE_DIM / 4; j += 32) dst[j] = src[j];
  }

  // ---- pass 2: recompute tiles, accumulate normalized soft assignments ----
  float mv[8], rv[8];
#pragma unroll
  for (int v = 0; v < 8; ++v) {
    mv[v] = s_m[wave][v + 8 * half];
    rv[v] = s_rinv[wave][v + 8 * half];
  }
  tile_loop([&](const v8f& acc, int n) {
    float een = s_enorm[n];
    float ps = 0.f;
#pragma unroll
    for (int v = 0; v < 8; ++v)
      ps += __expf(2.f * acc[v] - een - mv[v]) * rv[v];
    ps += __shfl_xor(ps, 16);          // both halves hit the same code n
    if (half == 0) atomicAdd(&s_avg[n], ps);
  });

  __syncthreads();
  for (int j = threadIdx.x; j < NUM_CODES; j += 128)
    atomicAdd(&g_avg[j], s_avg[j]);
}

// ---------------- final: entropies + scalars ----------------
__global__ void vq_final(const float* __restrict__ g_avg,
                         const float* __restrict__ g_bin,
                         const float* __restrict__ g_scal,
                         float* __restrict__ out) {
  __shared__ float r1[256], r2[256];
  float e1 = 0.f, e2 = 0.f;
  for (int j = threadIdx.x; j < NUM_CODES; j += 256) {
    float a = g_avg[j] * (1.0f / (float)BATCH);
    e1 += a * logf(a + 1e-10f);
    float q = g_bin[j] * (1.0f / (float)BATCH);
    e2 += q * logf(q + 1e-10f);
  }
  r1[threadIdx.x] = e1;
  r2[threadIdx.x] = e2;
  __syncthreads();
  for (int s = 128; s > 0; s >>= 1) {
    if (threadIdx.x < (unsigned)s) {
      r1[threadIdx.x] += r1[threadIdx.x + s];
      r2[threadIdx.x] += r2[threadIdx.x + s];
    }
    __syncthreads();
  }
  if (threadIdx.x == 0) {
    const size_t BD = (size_t)BATCH * CODE_DIM;
    float entropy = -r1[0];
    float diversity = logf((float)NUM_CODES) - entropy;
    float mse = g_scal[0] * (1.0f / ((float)BATCH * (float)CODE_DIM));
    float vq_loss = 0.25f * mse + mse + 0.1f * diversity;   // commit == embed fwd
    float perplexity = expf(-r2[0]);
    out[BD] = vq_loss;
    out[BD + 1] = perplexity;
    out[BD + 2 + BATCH] = diversity;
  }
}

extern "C" void kernel_launch(void* const* d_in, const int* in_sizes, int n_in,
                              void* d_out, int out_size, void* d_ws, size_t ws_size,
                              hipStream_t stream) {
  const float* z   = (const float*)d_in[0];   // [65536, 256] f32
  const float* emb = (const float*)d_in[1];   // [4096, 256] f32

  // workspace layout (~4.05 MB total)
  unsigned short* e_pack = (unsigned short*)d_ws;          // 4 MB packed hi|lo
  float* e_norm = (float*)(e_pack + (size_t)NUM_CODES * ROW_U16);
  float* g_avg  = e_norm + NUM_CODES;
  float* g_bin  = g_avg + NUM_CODES;
  float* g_scal = g_bin + NUM_CODES;   // 8 floats

  float* out = (float*)d_out;
  // out layout: [0,BD) z_q_st | [BD] vq_loss | [BD+1] perplexity |
  //             [BD+2, BD+2+B) indices (as f32) | [BD+2+B] diversity
  float* out_zq  = out;
  float* out_idx = out + (size_t)BATCH * CODE_DIM + 2;

  const int nzero = 2 * NUM_CODES + 8;
  vq_init<<<(nzero + 255) / 256, 256, 0, stream>>>(g_avg, nzero);
  vq_prep<<<NUM_CODES, CODE_DIM, 0, stream>>>(emb, e_pack, e_norm);
  vq_main<<<BATCH / 64, 128, 0, stream>>>(z, emb, e_pack, e_norm,
                                          g_avg, g_bin, g_scal, out_zq, out_idx);
  vq_final<<<1, 256, 0, stream>>>(g_avg, g_bin, g_scal, out);
}
